// SceneGuided_ConGNN_3118146257681
// MI455X (gfx1250) — matrix operand order repaired and checked by reading, not compiled, not run on hardware
//
#include <hip/hip_runtime.h>

// ---------- vector typedefs (avoid HIP struct vector types in fragments) ----
typedef __attribute__((ext_vector_type(16))) __bf16         bf16x16;
typedef __attribute__((ext_vector_type(16))) unsigned short u16x16;
typedef __attribute__((ext_vector_type(8)))  unsigned short u16x8;
typedef __attribute__((ext_vector_type(8)))  float          f32x8;

static __device__ __forceinline__ unsigned short f2bf(float f) {
  unsigned u = __float_as_uint(f);
  u += 0x7fffu + ((u >> 16) & 1u);      // round-to-nearest-even truncation
  return (unsigned short)(u >> 16);
}

// sortable-uint encoding for float atomic-max
static __device__ __forceinline__ unsigned fenc(float f) {
  unsigned u = __float_as_uint(f);
  return (u & 0x80000000u) ? ~u : (u | 0x80000000u);
}
static __device__ __forceinline__ float fdec(unsigned u) {
  return (u & 0x80000000u) ? __uint_as_float(u & 0x7fffffffu) : __uint_as_float(~u);
}
#define ENC_NEG_INF 0x007FFFFFu  // fenc(-inf)

// ---------- weight convert: W[K,N] f32 -> Wt[N,K] bf16 (transpose) ----------
__global__ __launch_bounds__(256) void wcvt_kernel(const float* __restrict__ W,
                                                   unsigned short* __restrict__ Wt,
                                                   int K, int N) {
  int t = blockIdx.x * 256 + threadIdx.x;
  if (t >= N * K) return;
  int n = t / K, k = t - n * K;
  Wt[t] = f2bf(W[(size_t)k * N + n]);
}

// ---------- activation convert: A[M,K] f32 -> bf16 (same layout, coalesced) -
__global__ __launch_bounds__(256) void acvt_kernel(const float* __restrict__ A,
                                                   unsigned short* __restrict__ Ab,
                                                   int tot) {
  int t = blockIdx.x * 256 + threadIdx.x;
  if (t < tot) Ab[t] = f2bf(A[t]);
}

// ---------- bf16 WMMA GEMM: C[M,N] = A[M,K] @ W[K,N] (+bias) ----------------
// A bf16 row-major [M][K]; Wt bf16 [N][K]; block = 8 waves; wave: 16r x 64c.
// Requires M%16==0, N%64==0, K%32==0 (all shapes here satisfy this).
// Inner loop: 2 A b128 loads + 8 B b128 loads + 4 v_wmma; zero conversion VALU.
__global__ __launch_bounds__(256) void gemm_bf16_kernel(
    const unsigned short* __restrict__ A, const unsigned short* __restrict__ Wt,
    const float* __restrict__ bias, float* __restrict__ C, int M, int N, int K) {
  const int lane = threadIdx.x & 31;
  const int wave = threadIdx.x >> 5;
  const int N0 = blockIdx.x * 64;
  const int Mw = blockIdx.y * 128 + wave * 16;
  if (Mw >= M) return;                       // wave-uniform: EXEC stays all-ones
  const int mrow = Mw + (lane & 15);
  const int kA = (lane >> 4) << 4;           // +0 / +16  (A half-lane K offset)
  const int kB = (lane >> 4) << 4;           // +0 / +16  (B half-lane K offset)
  const unsigned short* arow = A + (size_t)mrow * K;

  f32x8 acc0 = {0,0,0,0,0,0,0,0}, acc1 = acc0, acc2 = acc0, acc3 = acc0;

  for (int k0 = 0; k0 < K; k0 += 32) {
    if (k0 + 512 < K) __builtin_prefetch(arow + k0 + 512);   // global_prefetch_b8
    // ---- A fragment (contiguous, matches B): e -> k0+kA+e
    const u16x8* ap = (const u16x8*)(arow + k0 + kA);
    const u16x8* ap2 = (const u16x8*)(arow + k0 + 8 + kA);
    u16x8 al = ap[0], ah = ap2[0];
    u16x16 aw;
    aw[0]=al[0]; aw[1]=al[1]; aw[2]=al[2]; aw[3]=al[3];
    aw[4]=al[4]; aw[5]=al[5]; aw[6]=al[6]; aw[7]=al[7];
    aw[8]=ah[0]; aw[9]=ah[1]; aw[10]=ah[2]; aw[11]=ah[3];
    aw[12]=ah[4]; aw[13]=ah[5]; aw[14]=ah[6]; aw[15]=ah[7];
    bf16x16 av = __builtin_bit_cast(bf16x16, aw);

#define DO_TILE(J, ACC)                                                        \
    {                                                                          \
      const u16x8* wp = (const u16x8*)(Wt + (size_t)(N0 + (J)*16 + (lane & 15)) * K + k0 + kB); \
      u16x8 lo = wp[0], hi = wp[1];                                            \
      u16x16 bw;                                                               \
      bw[0]=lo[0]; bw[1]=lo[1]; bw[2]=lo[2]; bw[3]=lo[3];                      \
      bw[4]=lo[4]; bw[5]=lo[5]; bw[6]=lo[6]; bw[7]=lo[7];                      \
      bw[8]=hi[0]; bw[9]=hi[1]; bw[10]=hi[2]; bw[11]=hi[3];                    \
      bw[12]=hi[4]; bw[13]=hi[5]; bw[14]=hi[6]; bw[15]=hi[7];                  \
      bf16x16 bv = __builtin_bit_cast(bf16x16, bw);                            \
      ACC = __builtin_amdgcn_wmma_f32_16x16x32_bf16(false, av, false, bv,      \
                                                    (short)0, ACC, false, false); \
    }
    DO_TILE(0, acc0) DO_TILE(1, acc1) DO_TILE(2, acc2) DO_TILE(3, acc3)
#undef DO_TILE
  }

  // ---- epilogue: C/D layout: elem r -> row r+(lane>=16?8:0), col lane&15 ----
  const int mbase = Mw + ((lane >> 4) << 3);
#define WR_TILE(J, ACC)                                                        \
  {                                                                            \
    int nn = N0 + (J)*16 + (lane & 15);                                        \
    float bi = bias ? bias[nn] : 0.f;                                          \
    _Pragma("unroll")                                                          \
    for (int r = 0; r < 8; ++r) C[(size_t)(mbase + r) * N + nn] = ACC[r] + bi; \
  }
  WR_TILE(0, acc0) WR_TILE(1, acc1) WR_TILE(2, acc2) WR_TILE(3, acc3)
#undef WR_TILE
}

// ---------- row LayerNorm (+act, +residual, out row-stride) -----------------
// act: 0=none 1=relu 2=elu
__global__ __launch_bounds__(256) void ln_act_kernel(
    const float* __restrict__ X, const float* __restrict__ g,
    const float* __restrict__ b, const float* __restrict__ res,
    float* __restrict__ Out, int N, int ostride, int act) {
  __shared__ float r1[256], r2[256];
  const int row = blockIdx.x, tid = threadIdx.x;
  const float* xrow = X + (size_t)row * N;
  float xv[8]; int cnt = 0;
  float s = 0.f, s2 = 0.f;
  for (int i = tid; i < N; i += 256) { float v = xrow[i]; xv[cnt++] = v; s += v; s2 += v * v; }
  r1[tid] = s; r2[tid] = s2;
  __syncthreads();
  for (int st = 128; st > 0; st >>= 1) {
    if (tid < st) { r1[tid] += r1[tid + st]; r2[tid] += r2[tid + st]; }
    __syncthreads();
  }
  float mu = r1[0] / N;
  float var = r2[0] / N - mu * mu;
  float rstd = rsqrtf(var + 1e-5f);
  cnt = 0;
  for (int i = tid; i < N; i += 256) {
    float y = (xv[cnt++] - mu) * rstd * g[i] + b[i];
    if (act == 1) y = fmaxf(y, 0.f);
    else if (act == 2) y = (y > 0.f) ? y : (expf(y) - 1.f);
    if (res) y += res[(size_t)row * N + i];
    Out[(size_t)row * ostride + i] = y;
  }
}

// ---------- fills -----------------------------------------------------------
__global__ void fill_f32_kernel(float* p, float v, int n) {
  int i = blockIdx.x * 256 + threadIdx.x; if (i < n) p[i] = v;
}
__global__ void fill_u32_kernel(unsigned* p, unsigned v, int n) {
  int i = blockIdx.x * 256 + threadIdx.x; if (i < n) p[i] = v;
}

// ---------- GATv2 edge kernels (D=512, H=4, C=128); one wave per edge -------
__global__ __launch_bounds__(256) void edge_score_kernel(
    const int* __restrict__ src, const int* __restrict__ dst, int E, int Nn,
    const float* __restrict__ xl, const float* __restrict__ xr,
    const float* __restrict__ att, float* __restrict__ s, unsigned* __restrict__ mmax) {
  int e = blockIdx.x * 8 + (threadIdx.x >> 5);
  int lane = threadIdx.x & 31;
  if (e >= E + Nn) return;
  int si = (e < E) ? src[e] : (e - E);
  int di = (e < E) ? dst[e] : (e - E);
  const float* pl = xl + (size_t)si * 512 + lane * 16;
  const float* pr = xr + (size_t)di * 512 + lane * 16;
  const float* pa = att + lane * 16;
  float p = 0.f;
#pragma unroll
  for (int i = 0; i < 16; ++i) {
    float v = pl[i] + pr[i];
    v = (v > 0.f) ? v : 0.2f * v;        // leaky_relu(0.2)
    p += pa[i] * v;
  }
  p += __shfl_xor(p, 1); p += __shfl_xor(p, 2); p += __shfl_xor(p, 4);
  if ((lane & 7) == 0) {
    int h = lane >> 3;
    s[(size_t)e * 4 + h] = p;
    atomicMax(&mmax[(size_t)di * 4 + h], fenc(p));
  }
}

__global__ __launch_bounds__(256) void edge_exp_kernel(
    const int* __restrict__ dst, int E, int Nn,
    float* __restrict__ s, const unsigned* __restrict__ mmax, float* __restrict__ z) {
  int t = blockIdx.x * 256 + threadIdx.x;
  if (t >= (E + Nn) * 4) return;
  int e = t >> 2, h = t & 3;
  int di = (e < E) ? dst[e] : (e - E);
  float p = expf(s[t] - fdec(mmax[(size_t)di * 4 + h]));
  s[t] = p;
  atomicAdd(&z[(size_t)di * 4 + h], p);
}

__global__ __launch_bounds__(256) void edge_aggr_kernel(
    const int* __restrict__ src, const int* __restrict__ dst, int E, int Nn,
    const float* __restrict__ xl, const float* __restrict__ s,
    const float* __restrict__ z, float* __restrict__ hnew) {
  int e = blockIdx.x * 8 + (threadIdx.x >> 5);
  int lane = threadIdx.x & 31;
  if (e >= E + Nn) return;
  int si = (e < E) ? src[e] : (e - E);
  int di = (e < E) ? dst[e] : (e - E);
  int h = lane >> 3;
  float a = s[(size_t)e * 4 + h] / z[(size_t)di * 4 + h];
  const float* pl = xl + (size_t)si * 512 + lane * 16;
  float* po = hnew + (size_t)di * 512 + lane * 16;
#pragma unroll
  for (int i = 0; i < 16; ++i) atomicAdd(&po[i], a * pl[i]);
}

// ---------- fixed-size mean pool (per rows contiguous, D=512) ---------------
__global__ __launch_bounds__(256) void pool_kernel(const float* __restrict__ X,
                                                   float* __restrict__ P, int per) {
  int t = blockIdx.x * 256 + threadIdx.x;
  if (t >= 64 * 512) return;
  int b = t >> 9, d = t & 511;
  const float* xp = X + ((size_t)b * per) * 512 + d;
  float sum = 0.f;
  for (int r = 0; r < per; ++r) sum += xp[(size_t)r * 512];
  P[t] = sum / (float)per;
}

// ---------- tiny classifier: out[64,3] = X[64,512]@W[512,3]+b ---------------
__global__ void clf3_kernel(const float* __restrict__ X, const float* __restrict__ W,
                            const float* __restrict__ bias, float* __restrict__ out) {
  int t = threadIdx.x;
  if (t >= 192) return;
  int b = t / 3, c = t - b * 3;
  float acc = bias[c];
  const float* xp = X + (size_t)b * 512;
  for (int d = 0; d < 512; ++d) acc += xp[d] * W[d * 3 + c];
  out[t] = acc;
}

// ---------- elementwise helpers --------------------------------------------
__global__ void add_kernel(const float* a, const float* b, float* o, int n) {
  int i = blockIdx.x * 256 + threadIdx.x; if (i < n) o[i] = a[i] + b[i];
}
__global__ void addscale_kernel(const float* base, const float* t, const float* lam,
                                float* out, int ostride) {
  int i = blockIdx.x * 256 + threadIdx.x;
  if (i >= 64 * 512) return;
  int b = i >> 9, d = i & 511;
  out[(size_t)b * ostride + d] = base[i] + lam[0] * t[i];
}

// ---------- fusion: 1-query 4-head cross attention over 384 keys ------------
// dense batch is a fixed reindex: l<128 -> Hf[b*128+l], else Ho[b*256+l-128]
__global__ __launch_bounds__(256) void fusion_attn_kernel(
    const float* __restrict__ q, const float* __restrict__ Kf, const float* __restrict__ Ko,
    const float* __restrict__ Vf, const float* __restrict__ Vo, float* __restrict__ out) {
  __shared__ float sq[512];
  __shared__ float sc[4 * 384];
  int b = blockIdx.x, tid = threadIdx.x;
  for (int i = tid; i < 512; i += 256) sq[i] = q[(size_t)b * 512 + i];
  __syncthreads();
  for (int l = tid; l < 384; l += 256) {
    const float* kr = (l < 128) ? (Kf + ((size_t)b * 128 + l) * 512)
                                : (Ko + ((size_t)b * 256 + (l - 128)) * 512);
    for (int h = 0; h < 4; ++h) {
      float d = 0.f;
      for (int i = 0; i < 128; ++i) d += sq[h * 128 + i] * kr[h * 128 + i];
      sc[h * 384 + l] = d * 0.0883883476483184f;   // 1/sqrt(128)
    }
  }
  __syncthreads();
  if (tid < 4) {
    int h = tid;
    float mx = -1e30f;
    for (int l = 0; l < 384; ++l) mx = fmaxf(mx, sc[h * 384 + l]);
    float zs = 0.f;
    for (int l = 0; l < 384; ++l) { float p = expf(sc[h * 384 + l] - mx); sc[h * 384 + l] = p; zs += p; }
    float inv = 1.f / zs;
    for (int l = 0; l < 384; ++l) sc[h * 384 + l] *= inv;
  }
  __syncthreads();
  for (int t = tid; t < 512; t += 256) {
    int h = t >> 7, d = t & 127;
    float acc = 0.f;
    for (int l = 0; l < 384; ++l) {
      const float* vr = (l < 128) ? (Vf + ((size_t)b * 128 + l) * 512)
                                  : (Vo + ((size_t)b * 256 + (l - 128)) * 512);
      acc += sc[h * 384 + l] * vr[h * 128 + d];
    }
    out[(size_t)b * 512 + t] = acc;
  }
}

// ============================ host orchestration ============================
extern "C" void kernel_launch(void* const* d_in, const int* in_sizes, int n_in,
                              void* d_out, int out_size, void* d_ws, size_t ws_size,
                              hipStream_t stream) {
  (void)in_sizes; (void)n_in; (void)out_size; (void)ws_size;
  constexpr int NF = 8192, NO = 16384, EFc = 65536, EOc = 131072, B = 64;

  const float* face_x    = (const float*)d_in[0];
  const float* context_x = (const float*)d_in[1];
  const float* scene_x   = (const float*)d_in[2];
  const int*   fe        = (const int*)d_in[3];   // [2,EF]
  const int*   ce        = (const int*)d_in[4];   // [2,EO]
  auto P = [&](int i) { return (const float*)d_in[i]; };
  float* dout = (float*)d_out;

  // ---- workspace arena ----
  char* base = (char*)d_ws;
  size_t off = 0;
  auto alloc = [&](size_t bytes) -> void* {
    void* p = base + off;
    off += (bytes + 255) & ~(size_t)255;
    return p;
  };
  auto cvt = [&](int idx, int K, int N) -> unsigned short* {
    unsigned short* wt = (unsigned short*)alloc((size_t)N * K * 2);
    int tot = N * K;
    wcvt_kernel<<<(tot + 255) / 256, 256, 0, stream>>>(P(idx), wt, K, N);
    return wt;
  };

  // weights -> bf16 [N][K]
  unsigned short* rf_W1t = cvt(7, 4096, 1024);
  unsigned short* rf_W2t = cvt(11, 1024, 512);
  unsigned short* ro_Wt  = cvt(15, 2048, 512);
  unsigned short* rs_Wt  = cvt(19, 2048, 512);
  unsigned short* fg_inW = cvt(23, 512, 512);
  unsigned short* fg_Wl0 = cvt(27, 512, 512);
  unsigned short* fg_Wr0 = cvt(28, 512, 512);
  unsigned short* fg_Wl1 = cvt(32, 512, 512);
  unsigned short* fg_Wr1 = cvt(33, 512, 512);
  unsigned short* cg_inW = cvt(37, 512, 512);
  unsigned short* cg_Wl0 = cvt(41, 512, 512);
  unsigned short* cg_Wr0 = cvt(42, 512, 512);
  unsigned short* cg_Wl1 = cvt(46, 512, 512);
  unsigned short* cg_Wr1 = cvt(47, 512, 512);
  unsigned short* Wq_t   = cvt(57, 512, 512);
  unsigned short* Wk_t   = cvt(59, 512, 512);
  unsigned short* Wv_t   = cvt(61, 512, 512);
  unsigned short* Wo_t   = cvt(63, 512, 512);
  unsigned short* rfp_Wt = cvt(69, 512, 512);
  unsigned short* rop_Wt = cvt(71, 512, 512);
  unsigned short* cw_W1t = cvt(73, 1536, 512);

  // shared bf16 A scratch (stream-ordered reuse); max M*K = 8192*4096
  unsigned short* bfA = (unsigned short*)alloc((size_t)NF * 4096 * 2);  // 67 MB

  // activations / scratch (with reuse): big0 = fx1, later xl, later Kf|Vf
  float* big0 = (float*)alloc((size_t)NF * 1024 * 4);        // 33.5 MB
  float* fx   = (float*)alloc((size_t)NF * 512 * 4);
  float* ox   = (float*)alloc((size_t)NO * 512 * 4);
  float* sx   = (float*)alloc((size_t)B * 512 * 4);
  float* Hf   = (float*)alloc((size_t)NF * 512 * 4);
  float* Ho   = (float*)alloc((size_t)NO * 512 * 4);
  float* xrb  = (float*)alloc((size_t)NO * 512 * 4);         // xr, later Ko
  float* hnew = (float*)alloc((size_t)NO * 512 * 4);         // hnew, later Vo
  float* sbuf = (float*)alloc((size_t)(EOc + NO) * 4 * 4);
  unsigned* mbuf = (unsigned*)alloc((size_t)NO * 4 * 4);
  float* zbuf = (float*)alloc((size_t)NO * 4 * 4);
  float* gfp  = (float*)alloc((size_t)B * 512 * 4);
  float* gop  = (float*)alloc((size_t)B * 512 * 4);
  float* fxp  = (float*)alloc((size_t)B * 512 * 4);
  float* oxp  = (float*)alloc((size_t)B * 512 * 4);
  float* qbuf = (float*)alloc((size_t)B * 512 * 4);
  float* attnb= (float*)alloc((size_t)B * 512 * 4);
  float* aob  = (float*)alloc((size_t)B * 512 * 4);
  float* t1   = (float*)alloc((size_t)B * 512 * 4);
  float* t2   = (float*)alloc((size_t)B * 512 * 4);
  float* hwpre= (float*)alloc((size_t)B * 512 * 4);
  float* hw   = (float*)alloc((size_t)B * 512 * 4);
  float* comb = (float*)alloc((size_t)B * 1536 * 4);
  float* fx1 = big0;

  auto tobf = [&](const float* A, int M, int K) {
    int tot = M * K;
    acvt_kernel<<<(tot + 255) / 256, 256, 0, stream>>>(A, bfA, tot);
  };
  auto gemmb = [&](const unsigned short* Wt, const float* bias,
                   float* C, int M, int N, int K) {
    dim3 g(N / 64, (M + 127) / 128);
    gemm_bf16_kernel<<<g, 256, 0, stream>>>(bfA, Wt, bias, C, M, N, K);
  };
  auto ln = [&](const float* X, const float* g, const float* b, const float* res,
                float* Out, int M, int N, int ostride, int act) {
    ln_act_kernel<<<M, 256, 0, stream>>>(X, g, b, res, Out, N, ostride, act);
  };

  // ---- feature reducers ----
  tobf(face_x, NF, 4096);
  gemmb(rf_W1t, P(8), fx1, NF, 1024, 4096);
  ln(fx1, P(9), P(10), nullptr, fx1, NF, 1024, 1024, 1);
  tobf(fx1, NF, 1024);
  gemmb(rf_W2t, P(12), fx, NF, 512, 1024);
  ln(fx, P(13), P(14), nullptr, fx, NF, 512, 512, 1);
  tobf(context_x, NO, 2048);
  gemmb(ro_Wt, P(16), ox, NO, 512, 2048);
  ln(ox, P(17), P(18), nullptr, ox, NO, 512, 512, 1);
  tobf(scene_x, B, 2048);
  gemmb(rs_Wt, P(20), sx, B, 512, 2048);
  ln(sx, P(21), P(22), nullptr, sx, B, 512, 512, 1);

  // ---- GATv2 layer (h converted once, used by both Wl and Wr GEMMs) ----
  auto gat_layer = [&](float* h, int Nn, const int* esrc, const int* edst, int E,
                       const unsigned short* Wlt, const unsigned short* Wrt,
                       const float* att, const float* lg, const float* lb) {
    float* xl = big0;
    tobf(h, Nn, 512);
    gemmb(Wlt, nullptr, xl, Nn, 512, 512);
    gemmb(Wrt, nullptr, xrb, Nn, 512, 512);
    fill_u32_kernel<<<(Nn * 4 + 255) / 256, 256, 0, stream>>>(mbuf, ENC_NEG_INF, Nn * 4);
    fill_f32_kernel<<<(Nn * 4 + 255) / 256, 256, 0, stream>>>(zbuf, 0.f, Nn * 4);
    fill_f32_kernel<<<(Nn * 512 + 255) / 256, 256, 0, stream>>>(hnew, 0.f, Nn * 512);
    int tot = E + Nn;
    edge_score_kernel<<<(tot + 7) / 8, 256, 0, stream>>>(esrc, edst, E, Nn, xl, xrb, att, sbuf, mbuf);
    edge_exp_kernel<<<(tot * 4 + 255) / 256, 256, 0, stream>>>(edst, E, Nn, sbuf, mbuf, zbuf);
    edge_aggr_kernel<<<(tot + 7) / 8, 256, 0, stream>>>(esrc, edst, E, Nn, xl, sbuf, zbuf, hnew);
    ln(hnew, lg, lb, h, h, Nn, 512, 512, 2);     // h += elu(LN(hnew))
  };

  // face GAT
  tobf(fx, NF, 512);
  gemmb(fg_inW, P(24), Hf, NF, 512, 512);
  ln(Hf, P(25), P(26), nullptr, Hf, NF, 512, 512, 1);
  gat_layer(Hf, NF, fe, fe + EFc, EFc, fg_Wl0, fg_Wr0, P(29), P(30), P(31));
  gat_layer(Hf, NF, fe, fe + EFc, EFc, fg_Wl1, fg_Wr1, P(34), P(35), P(36));
  // context GAT
  tobf(ox, NO, 512);
  gemmb(cg_inW, P(38), Ho, NO, 512, 512);
  ln(Ho, P(39), P(40), nullptr, Ho, NO, 512, 512, 1);
  gat_layer(Ho, NO, ce, ce + EOc, EOc, cg_Wl0, cg_Wr0, P(43), P(44), P(45));
  gat_layer(Ho, NO, ce, ce + EOc, EOc, cg_Wl1, cg_Wr1, P(48), P(49), P(50));

  // ---- pooled features (fixed 128/256 rows per batch) ----
  pool_kernel<<<(B * 512 + 255) / 256, 256, 0, stream>>>(Hf, gfp, 128);
  pool_kernel<<<(B * 512 + 255) / 256, 256, 0, stream>>>(Ho, gop, 256);
  pool_kernel<<<(B * 512 + 255) / 256, 256, 0, stream>>>(fx, fxp, 128);
  pool_kernel<<<(B * 512 + 255) / 256, 256, 0, stream>>>(ox, oxp, 256);

  // ---- heads ----
  clf3_kernel<<<1, 192, 0, stream>>>(gfp, P(51), P(52), dout + 0);    // out_face
  clf3_kernel<<<1, 192, 0, stream>>>(gop, P(53), P(54), dout + 192);  // out_context
  clf3_kernel<<<1, 192, 0, stream>>>(sx,  P(55), P(56), dout + 384);  // out_scene

  // ---- fusion cross-attention (Hf/Ho each converted once for K and V) ----
  float* Kf = big0;                          // reuse (xl no longer needed)
  float* Vf = big0 + (size_t)NF * 512;
  float* Ko = xrb;
  float* Vo = hnew;
  tobf(sx, B, 512);
  gemmb(Wq_t, P(58), qbuf, B, 512, 512);
  tobf(Hf, NF, 512);
  gemmb(Wk_t, P(60), Kf, NF, 512, 512);
  gemmb(Wv_t, P(62), Vf, NF, 512, 512);
  tobf(Ho, NO, 512);
  gemmb(Wk_t, P(60), Ko, NO, 512, 512);
  gemmb(Wv_t, P(62), Vo, NO, 512, 512);
  fusion_attn_kernel<<<B, 256, 0, stream>>>(qbuf, Kf, Ko, Vf, Vo, attnb);
  tobf(attnb, B, 512);
  gemmb(Wo_t, P(64), aob, B, 512, 512);
  add_kernel<<<(B * 512 + 255) / 256, 256, 0, stream>>>(sx, aob, t2, B * 512);
  ln(t2, P(65), P(66), nullptr, comb, B, 512, 1536, 0);               // fused -> comb[:,0:512]

  // ---- feat_face / feat_obj into comb columns ----
  tobf(fxp, B, 512);
  gemmb(rfp_Wt, P(70), t1, B, 512, 512);
  addscale_kernel<<<(B * 512 + 255) / 256, 256, 0, stream>>>(gfp, t1, P(67), comb + 512, 1536);
  tobf(oxp, B, 512);
  gemmb(rop_Wt, P(72), t1, B, 512, 512);
  addscale_kernel<<<(B * 512 + 255) / 256, 256, 0, stream>>>(gop, t1, P(68), comb + 1024, 1536);

  // ---- whole-scene head ----
  tobf(comb, B, 1536);
  gemmb(cw_W1t, P(74), hwpre, B, 512, 1536);
  ln(hwpre, P(75), P(76), nullptr, hw, B, 512, 512, 1);
  clf3_kernel<<<1, 192, 0, stream>>>(hw, P(77), P(78), dout + 576);   // out_whole
}
